// Qwen3Attention_601295421920
// MI455X (gfx1250) — compile-verified, matrix operand order checked by reading, not run on hardware
//
#include <hip/hip_runtime.h>
#include <hip/hip_bf16.h>
#include <math.h>
#include <stdint.h>

#define T_SEQ   2048
#define HID     4096
#define NHEADS  32
#define NKV     8
#define HDIM    128
#define KVD     (NKV * HDIM)      // 1024
#define GA_LD   40                // padded LDS row stride (elements, 80B = 16B-aligned)

typedef __bf16 bf16_t;
typedef unsigned int u32;
typedef __attribute__((ext_vector_type(16))) __bf16 v16bf;
typedef __attribute__((ext_vector_type(8)))  __bf16 v8bf;
typedef __attribute__((ext_vector_type(8)))  float  v8f;
typedef __attribute__((ext_vector_type(4)))  u32    v4u;
typedef __attribute__((ext_vector_type(8)))  u32    v8u;

__device__ __forceinline__ v16bf mk16(v8bf lo, v8bf hi) {
  v16bf r;
#pragma unroll
  for (int i = 0; i < 8; ++i) { r[i] = lo[i]; r[i + 8] = hi[i]; }
  return r;
}

__device__ __forceinline__ v8f zero8() {
  v8f z;
#pragma unroll
  for (int i = 0; i < 8; ++i) z[i] = 0.0f;
  return z;
}

__device__ __forceinline__ v8f wmma32(v16bf a, v16bf b, v8f c) {
  // D = A(16x32 bf16) * B(32x16 bf16) + C(16x16 f32)
  return __builtin_amdgcn_wmma_f32_16x16x32_bf16(false, a, false, b, (short)0, c,
                                                 false, false);
}

// ---------------------------------------------------------------------------
// Tensor Data Mover: 2-D tile load global -> LDS (D# built per ISA ch.8).
// data_size = 2 bytes. Optional LDS padding (pad per tile row) for bank spread.
// Issued redundantly by every wave (idempotent writes); EXEC is ignored by TDM.
// ---------------------------------------------------------------------------
__device__ __forceinline__ void tdm_load_2d(void* lds_dst, const void* gsrc,
                                            u32 tensor_w, u32 tensor_h,
                                            u32 tile_w, u32 tile_h,
                                            u32 stride_elems,
                                            u32 pad_en, u32 pad_interval,
                                            u32 pad_amount) {
  unsigned long long ga = (unsigned long long)(size_t)gsrc;
  u32 lds_off = (u32)(size_t)lds_dst;  // LDS aperture: addr[31:0] = LDS address
  v4u g0;
  g0[0] = 1u;                               // count=1, user flags 0
  g0[1] = lds_off;                          // lds_addr
  g0[2] = (u32)ga;                          // global_addr[31:0]
  g0[3] = (u32)(ga >> 32) | (2u << 30);     // global_addr[56:32], type=2 (image)
  v8u g1;
  g1[0] = (1u << 16)                        // data_size = 1 -> 2 bytes
        | (pad_en << 20) | (pad_interval << 22) | (pad_amount << 25);
  g1[1] = (tensor_w & 0xFFFFu) << 16;                         // tensor_dim0 lo
  g1[2] = (tensor_w >> 16) | ((tensor_h & 0xFFFFu) << 16);    // dim0 hi, dim1 lo
  g1[3] = (tensor_h >> 16) | ((tile_w & 0xFFFFu) << 16);      // dim1 hi, tile_dim0
  g1[4] = tile_h & 0xFFFFu;                                   // tile_dim1 (tile_dim2=0)
  g1[5] = stride_elems;                                       // tensor_dim0_stride lo
  g1[6] = 0u;
  g1[7] = 0u;
  asm volatile("tensor_load_to_lds %0, %1" :: "s"(g0), "s"(g1) : "memory");
}

__device__ __forceinline__ void tdm_wait() {
  __builtin_amdgcn_s_wait_tensorcnt(0);
}

// ---------------------------------------------------------------------------
// f32 -> bf16 cast (grid-stride)
// ---------------------------------------------------------------------------
__global__ void k_cast_bf16(const float* __restrict__ s, bf16_t* __restrict__ d,
                            long n) {
  long i = (long)blockIdx.x * blockDim.x + threadIdx.x;
  long str = (long)gridDim.x * blockDim.x;
  for (; i < n; i += str) d[i] = (bf16_t)s[i];
}

// ---------------------------------------------------------------------------
// GEMM: C(f32, MxN) = A(bf16, MxK row-major) * B(bf16, KxN row-major)
// 256 threads = 8 waves; block tile 128x128; K-step 32; wave tile 32x64.
// A tile staged by TDM (hardware-padded rows: 64B data + 16B pad = 40 elems).
// ---------------------------------------------------------------------------
__global__ __launch_bounds__(256) void k_gemm_bf16(
    const bf16_t* __restrict__ A, const bf16_t* __restrict__ B,
    float* __restrict__ C, int M, int N, int K) {
  __shared__ bf16_t As[128 * GA_LD];   // [m][k], padded by TDM
  __shared__ bf16_t Bs[128 * GA_LD];   // transposed: [n][k]
  const int tid = threadIdx.x;
  const int wave = tid >> 5, lane = tid & 31;
  const int lr = lane & 15;
  const int klo = (lane >> 4) * 8;
  const int bm = blockIdx.x * 128;
  const int bn = blockIdx.y * 128;
  const int wm = (wave & 3) * 32;
  const int wn = (wave >> 2) * 64;

  v8f acc[2][4];
#pragma unroll
  for (int i = 0; i < 2; ++i)
#pragma unroll
    for (int j = 0; j < 4; ++j) acc[i][j] = zero8();

  for (int k0 = 0; k0 < K; k0 += 32) {
    // A tile 128x32 via Tensor Data Mover; pad_interval=3 (16 DW = 64B row),
    // pad_amount=3 (4 DW = 16B) -> LDS row stride 80B = GA_LD elements.
    tdm_load_2d(As, A + (long)bm * K + k0, (u32)K, (u32)M, 32u, 128u, (u32)K,
                1u, 3u, 3u);
    {  // stage B tile 32x128, transposed into Bs[n][k]
      int kk = tid >> 3, n0 = (tid & 7) * 16;
      const bf16_t* src = B + (long)(k0 + kk) * N + bn + n0;
      v8bf b0 = *(const v8bf*)(src);
      v8bf b1 = *(const v8bf*)(src + 8);
#pragma unroll
      for (int i = 0; i < 8; ++i) Bs[(n0 + i) * GA_LD + kk] = b0[i];
#pragma unroll
      for (int i = 0; i < 8; ++i) Bs[(n0 + 8 + i) * GA_LD + kk] = b1[i];
      if (k0 + 32 < K)  // prefetch next B tile row into caches
        __builtin_prefetch(src + 32 * (long)N, 0, 1);
    }
    tdm_wait();
    __syncthreads();

    v16bf af[2], bfr[4];
#pragma unroll
    for (int i = 0; i < 2; ++i) {
      const bf16_t* p = &As[(wm + i * 16 + lr) * GA_LD];
      af[i] = mk16(*(const v8bf*)(p + klo), *(const v8bf*)(p + 16 + klo));
    }
#pragma unroll
    for (int j = 0; j < 4; ++j) {
      const bf16_t* p = &Bs[(wn + j * 16 + lr) * GA_LD];
      bfr[j] = mk16(*(const v8bf*)(p + klo), *(const v8bf*)(p + 16 + klo));
    }
#pragma unroll
    for (int i = 0; i < 2; ++i)
#pragma unroll
      for (int j = 0; j < 4; ++j) acc[i][j] = wmma32(af[i], bfr[j], acc[i][j]);
    __syncthreads();
  }

  // D layout: vgpr r -> row (r + 8*(lane/16)), col lane%16
#pragma unroll
  for (int i = 0; i < 2; ++i)
#pragma unroll
    for (int j = 0; j < 4; ++j)
#pragma unroll
      for (int r = 0; r < 8; ++r) {
        int m = bm + wm + i * 16 + r + (lane >> 4) * 8;
        int n = bn + wn + j * 16 + lr;
        C[(long)m * N + n] = acc[i][j][r];
      }
}

// ---------------------------------------------------------------------------
// Fused RMSNorm + RoPE, f32 in -> bf16 out. grid (T, heads), block 128.
// ---------------------------------------------------------------------------
__global__ __launch_bounds__(128) void k_norm_rope(
    const float* __restrict__ X, const int* __restrict__ pos,
    const float* __restrict__ w, bf16_t* __restrict__ Y, int rowstride) {
  const int t = blockIdx.x, h = blockIdx.y, i = threadIdx.x;
  __shared__ float xs[HDIM];
  __shared__ float red[4];
  const long base = (long)t * rowstride + h * HDIM;
  float x = X[base + i];
  float ss = x * x;
#pragma unroll
  for (int off = 1; off < 32; off <<= 1) ss += __shfl_xor(ss, off);
  if ((i & 31) == 0) red[i >> 5] = ss;
  __syncthreads();
  float rms = rsqrtf((red[0] + red[1] + red[2] + red[3]) * (1.0f / HDIM) + 1e-6f);
  float xn = x * rms * w[i];
  xs[i] = xn;
  __syncthreads();
  const int j = i & 63;
  // inv_freq = 10000^(-2j/128); ln(10000) = 9.210340372
  float invf = __expf(-((float)(2 * j) * (1.0f / HDIM)) * 9.2103403719761836f);
  float fr = (float)pos[t] * invf;
  float sn, cs;
  __sincosf(fr, &sn, &cs);
  float partner = xs[(i < 64) ? i + 64 : i - 64];
  float o = (i < 64) ? (xn * cs - partner * sn) : (xn * cs + partner * sn);
  Y[base + i] = (bf16_t)o;
}

// ---------------------------------------------------------------------------
// Causal flash attention. grid (T/128, NHEADS), block 256 (8 waves).
// Each wave owns 16 query rows x full head_dim(128); kv tiles of 32 keys.
// K tile staged by TDM; V tile staged transposed for B-fragment reads.
// ---------------------------------------------------------------------------
__global__ __launch_bounds__(256) void k_flash(
    const bf16_t* __restrict__ Q, const bf16_t* __restrict__ Kg,
    const bf16_t* __restrict__ Vg, bf16_t* __restrict__ O) {
  __shared__ bf16_t Ks[32 * 128];       // [key][d]  (natural, via TDM)
  __shared__ bf16_t Vs[128 * GA_LD];    // [d][key]  (transposed, padded)
  __shared__ bf16_t Ps[8][16 * 32];     // per-wave probability tile
  const int tid = threadIdx.x, wave = tid >> 5, lane = tid & 31;
  const int lr = lane & 15, klo = (lane >> 4) * 8;
  const int h = blockIdx.y, hk = h >> 2;       // GQA: 4 q-heads per kv-head
  const int q0 = blockIdx.x * 128;
  const int wq = q0 + wave * 16;
  const float scale = 0.088388347648318447f;   // 1/sqrt(128)

  // Q fragments for this wave (reused across all kv tiles)
  v16bf qf[4];
#pragma unroll
  for (int kk = 0; kk < 4; ++kk) {
    const bf16_t* p = Q + (long)(wq + lr) * (NHEADS * HDIM) + h * HDIM + kk * 32;
    qf[kk] = mk16(*(const v8bf*)(p + klo), *(const v8bf*)(p + 16 + klo));
  }
  v8f oacc[8];
#pragma unroll
  for (int j = 0; j < 8; ++j) oacc[j] = zero8();
  float mrow[8], lrow[8];
#pragma unroll
  for (int r = 0; r < 8; ++r) { mrow[r] = -3.0e38f; lrow[r] = 0.0f; }

  const int nkt = (q0 >> 5) + 4;  // cover keys up to last query row in block
  for (int kt = 0; kt < nkt; ++kt) {
    const int kbase = kt * 32;
    // K tile 32 keys x 128 d via TDM (natural layout feeds B-fragments)
    tdm_load_2d(Ks, Kg + (long)kbase * KVD + hk * HDIM, 128u, (u32)T_SEQ,
                128u, 32u, (u32)KVD, 0u, 0u, 0u);
    {  // stage V transposed: Vs[d][key]
      int kr = tid >> 3, c = (tid & 7) * 16;
      const bf16_t* vsrc = Vg + (long)(kbase + kr) * KVD + hk * HDIM + c;
      v8bf v0 = *(const v8bf*)(vsrc);
      v8bf v1 = *(const v8bf*)(vsrc + 8);
#pragma unroll
      for (int i = 0; i < 8; ++i) Vs[(c + i) * GA_LD + kr] = v0[i];
#pragma unroll
      for (int i = 0; i < 8; ++i) Vs[(c + 8 + i) * GA_LD + kr] = v1[i];
      if (kt + 1 < nkt) {
        __builtin_prefetch(vsrc + 32 * (long)KVD, 0, 1);
        __builtin_prefetch(Kg + (long)(kbase + 32 + kr) * KVD + hk * HDIM + c, 0, 1);
      }
    }
    tdm_wait();
    __syncthreads();

    // scores: two 16x16 tiles (keys kbase..+15 and +16..+31), contract d=128
    v8f s0 = zero8(), s1 = zero8();
#pragma unroll
    for (int kk = 0; kk < 4; ++kk) {
      const bf16_t* p0 = &Ks[lr * 128 + kk * 32];
      const bf16_t* p1 = &Ks[(lr + 16) * 128 + kk * 32];
      v16bf b0 = mk16(*(const v8bf*)(p0 + klo), *(const v8bf*)(p0 + 16 + klo));
      v16bf b1 = mk16(*(const v8bf*)(p1 + klo), *(const v8bf*)(p1 + 16 + klo));
      s0 = wmma32(qf[kk], b0, s0);
      s1 = wmma32(qf[kk], b1, s1);
    }

    // online softmax (row stats replicated across each 16-lane group)
#pragma unroll
    for (int r = 0; r < 8; ++r) {
      const int m = wq + r + (lane >> 4) * 8;   // global query row
      float a = (kbase + lr      <= m) ? s0[r] * scale : -3.0e38f;
      float b = (kbase + lr + 16 <= m) ? s1[r] * scale : -3.0e38f;
      float mx = fmaxf(a, b);
#pragma unroll
      for (int off = 1; off < 16; off <<= 1) mx = fmaxf(mx, __shfl_xor(mx, off));
      float mnew = fmaxf(mrow[r], mx);
      float corr = __expf(mrow[r] - mnew);
      float p0 = __expf(a - mnew);
      float p1 = __expf(b - mnew);
      float psum = p0 + p1;
#pragma unroll
      for (int off = 1; off < 16; off <<= 1) psum += __shfl_xor(psum, off);
      lrow[r] = lrow[r] * corr + psum;
      mrow[r] = mnew;
#pragma unroll
      for (int j = 0; j < 8; ++j) oacc[j][r] *= corr;
      const int pm = r + (lane >> 4) * 8;       // D-layout -> A-layout via LDS
      Ps[wave][pm * 32 + lr]      = (bf16_t)p0;
      Ps[wave][pm * 32 + lr + 16] = (bf16_t)p1;
    }
    __syncthreads();

    // P (16x32, A-matrix) x V (32x128, B-matrix) -> accumulate into oacc
    const bf16_t* pp = &Ps[wave][lr * 32];
    v16bf pf = mk16(*(const v8bf*)(pp + klo), *(const v8bf*)(pp + 16 + klo));
#pragma unroll
    for (int j = 0; j < 8; ++j) {
      const bf16_t* vp = &Vs[(j * 16 + lr) * GA_LD];
      v16bf vfr = mk16(*(const v8bf*)(vp + klo), *(const v8bf*)(vp + 16 + klo));
      oacc[j] = wmma32(pf, vfr, oacc[j]);
    }
    __syncthreads();
  }

#pragma unroll
  for (int j = 0; j < 8; ++j)
#pragma unroll
    for (int r = 0; r < 8; ++r) {
      int m = wq + r + (lane >> 4) * 8;
      int d = j * 16 + lr;
      O[(long)m * (NHEADS * HDIM) + h * HDIM + d] = (bf16_t)(oacc[j][r] / lrow[r]);
    }
}

// ---------------------------------------------------------------------------
extern "C" void kernel_launch(void* const* d_in, const int* in_sizes, int n_in,
                              void* d_out, int out_size, void* d_ws, size_t ws_size,
                              hipStream_t stream) {
  (void)in_sizes; (void)n_in; (void)out_size; (void)ws_size;
  const int*   pos = (const int*)d_in[0];
  const float* hs  = (const float*)d_in[1];
  const float* wq  = (const float*)d_in[2];
  const float* wk  = (const float*)d_in[3];
  const float* wv  = (const float*)d_in[4];
  const float* wo  = (const float*)d_in[5];
  const float* qnw = (const float*)d_in[6];
  const float* knw = (const float*)d_in[7];
  float* out = (float*)d_out;

  char* ws = (char*)d_ws;
  size_t off = 0;
  auto carve = [&](size_t bytes) -> char* {
    char* p = ws + off;
    off += (bytes + 255) & ~(size_t)255;
    return p;
  };
  bf16_t* hsb   = (bf16_t*)carve((size_t)T_SEQ * HID * 2);
  bf16_t* wqb   = (bf16_t*)carve((size_t)HID * HID * 2);
  bf16_t* wkb   = (bf16_t*)carve((size_t)HID * KVD * 2);
  bf16_t* wvb   = (bf16_t*)carve((size_t)HID * KVD * 2);
  bf16_t* wob   = (bf16_t*)carve((size_t)HID * HID * 2);
  float*  qf    = (float*)carve((size_t)T_SEQ * HID * 4);
  float*  kf    = (float*)carve((size_t)T_SEQ * KVD * 4);
  float*  vf    = (float*)carve((size_t)T_SEQ * KVD * 4);
  bf16_t* qb    = (bf16_t*)carve((size_t)T_SEQ * HID * 2);
  bf16_t* kb    = (bf16_t*)carve((size_t)T_SEQ * KVD * 2);
  bf16_t* vb    = (bf16_t*)carve((size_t)T_SEQ * KVD * 2);
  bf16_t* attnb = (bf16_t*)carve((size_t)T_SEQ * HID * 2);

  // 1) casts to bf16
  k_cast_bf16<<<4096, 256, 0, stream>>>(hs, hsb, (long)T_SEQ * HID);
  k_cast_bf16<<<8192, 256, 0, stream>>>(wq, wqb, (long)HID * HID);
  k_cast_bf16<<<4096, 256, 0, stream>>>(wk, wkb, (long)HID * KVD);
  k_cast_bf16<<<4096, 256, 0, stream>>>(wv, wvb, (long)HID * KVD);
  k_cast_bf16<<<8192, 256, 0, stream>>>(wo, wob, (long)HID * HID);

  // 2) QKV projections (WMMA GEMM, TDM-staged A tiles)
  k_gemm_bf16<<<dim3(T_SEQ / 128, HID / 128), 256, 0, stream>>>(hsb, wqb, qf,
                                                                T_SEQ, HID, HID);
  k_gemm_bf16<<<dim3(T_SEQ / 128, KVD / 128), 256, 0, stream>>>(hsb, wkb, kf,
                                                                T_SEQ, KVD, HID);
  k_gemm_bf16<<<dim3(T_SEQ / 128, KVD / 128), 256, 0, stream>>>(hsb, wvb, vf,
                                                                T_SEQ, KVD, HID);

  // 3) rmsnorm + rope (q, k); cast v
  k_norm_rope<<<dim3(T_SEQ, NHEADS), 128, 0, stream>>>(qf, pos, qnw, qb, HID);
  k_norm_rope<<<dim3(T_SEQ, NKV), 128, 0, stream>>>(kf, pos, knw, kb, KVD);
  k_cast_bf16<<<2048, 256, 0, stream>>>(vf, vb, (long)T_SEQ * KVD);

  // 4) causal flash attention
  k_flash<<<dim3(T_SEQ / 128, NHEADS), 256, 0, stream>>>(qb, kb, vb, attnb);

  // 5) output projection -> f32 d_out
  k_gemm_bf16<<<dim3(T_SEQ / 128, HID / 128), 256, 0, stream>>>(attnb, wob, out,
                                                                T_SEQ, HID, HID);
}
